// MultiHeadAttentionNaive_68169720922468
// MI455X (gfx1250) — compile-verified, matrix operand order checked by reading, not run on hardware
//
#include <hip/hip_runtime.h>

#define D_MODEL  1024
#define NUM_HEADS 16
#define HEAD_DIM  64
#define SEQ      1024
#define BATCH       8

typedef __attribute__((ext_vector_type(16))) __bf16 bf16x16;
typedef __attribute__((ext_vector_type(8)))  float  v8f;
typedef __attribute__((ext_vector_type(4)))  unsigned int u32x4;
typedef __attribute__((ext_vector_type(8)))  int i32x8;
typedef __attribute__((ext_vector_type(4)))  int i32x4;

union FragB16 { uint4 u[2]; bf16x16 v; };

// round-to-nearest-even f32 -> bf16 (bit form)
__device__ __forceinline__ unsigned short to_bf16(float f) {
    unsigned int u = __float_as_uint(f);
    u += 0x7FFFu + ((u >> 16) & 1u);
    return (unsigned short)(u >> 16);
}

// LDS aperture: flat address low 32 bits == LDS byte offset (ISA 10.2 aperture table)
__device__ __forceinline__ unsigned lds_off(const void* p) {
    return (unsigned)(size_t)p;
}

// ---------------------------------------------------------------------------
// TDM: DMA a 2D bf16 tile (tile_k contiguous elems x tile_rows rows) from
// global to LDS, padding each 16-DWORD row with 4 DWORDs (LDA=40 layout).
// D# packing per CDNA5 ISA (6-arg builtin).
// ---------------------------------------------------------------------------
__device__ __forceinline__ void tdm_load_tile_2d(
    const unsigned short* gsrc, unsigned lds_byte_off,
    unsigned tile_k, unsigned tile_rows,
    unsigned tensor_k, unsigned tensor_rows, unsigned row_stride)
{
    unsigned long long ga = (unsigned long long)(size_t)gsrc;
    u32x4 g0;
    g0[0] = 1u;                                                // count=1, user mode
    g0[1] = lds_byte_off;                                      // lds_addr
    g0[2] = (unsigned)ga;                                      // global_addr[31:0]
    g0[3] = (unsigned)((ga >> 32) & 0x01FFFFFFu) | (2u << 30); // addr[56:32] | type=2
    i32x8 g1;
    // data_size=1 (2B); pad_enable; pad_interval=3 (16 DW); pad_amount=3 (4 DW)
    g1[0] = (int)((1u << 16) | (1u << 20) | (3u << 22) | (3u << 25));
    g1[1] = (int)((tensor_k & 0xFFFFu) << 16);                 // tensor_dim0 lo16
    g1[2] = (int)(((tensor_k >> 16) & 0xFFFFu) | ((tensor_rows & 0xFFFFu) << 16));
    g1[3] = (int)(((tensor_rows >> 16) & 0xFFFFu) | ((tile_k & 0xFFFFu) << 16));
    g1[4] = (int)(tile_rows & 0xFFFFu);                        // tile_dim1; tile_dim2=0
    g1[5] = (int)row_stride;                                   // tensor_dim0_stride lo32
    g1[6] = 0;
    g1[7] = 0;
    i32x4 z4; z4[0] = z4[1] = z4[2] = z4[3] = 0;               // 2D: groups 2/3 unused
    i32x8 z8;
#pragma unroll
    for (int i = 0; i < 8; i++) z8[i] = 0;
    __builtin_amdgcn_tensor_load_to_lds(g0, g1, z4, z4, z8, 0);
}

// ---------------------------------------------------------------------------
// One-time prep: W f32 [K,N] -> Wt bf16 [N,K] (transposed) so GEMM B-tiles are
// contiguous-K and TDM-able.  grid(32,32), block 256.
// ---------------------------------------------------------------------------
__global__ __launch_bounds__(256) void transpose_w_bf16(
    const float* __restrict__ W, unsigned short* __restrict__ Wt)
{
    __shared__ float t[32][33];
    const int n0 = blockIdx.x * 32, k0 = blockIdx.y * 32;
    const int tx = threadIdx.x & 31, ty = threadIdx.x >> 5;
#pragma unroll
    for (int r = ty; r < 32; r += 8)
        t[r][tx] = W[(size_t)(k0 + r) * D_MODEL + n0 + tx];
    __syncthreads();
#pragma unroll
    for (int r = ty; r < 32; r += 8)
        Wt[(size_t)(n0 + r) * D_MODEL + k0 + tx] = to_bf16(t[tx][r]);
}

// ---------------------------------------------------------------------------
// GEMM: Y = X @ W + bias with W given as Wt bf16 [N,K].
// Software-pipelined, double-buffered: TDM for tile s+1 issued right after the
// single per-step barrier (DMA overlaps all 8 WMMAs); f32 A data for s+1 is
// fetched to registers before compute and stored to LDS after.
// Block tile 128x128, 8 waves (4x2), wave tile 32x64 = 2x4 WMMA frags.
// ---------------------------------------------------------------------------
template <typename TIn, int OUT_QKV>
__global__ __launch_bounds__(256) void gemm_bf16_wmma(
    const TIn* __restrict__ X, const unsigned short* __restrict__ Wt,
    const float* __restrict__ bias, void* __restrict__ out)
{
    constexpr int K = 1024, N = 1024, LDA = 40, NSTEP = K / 32;
    __shared__ unsigned short lds_a[2][128 * LDA];   // A tile [r][k]
    __shared__ unsigned short lds_b[2][128 * LDA];   // B tile [n][k]

    const int m0 = blockIdx.x * 128, n0 = blockIdx.y * 128;
    const int tid  = threadIdx.x;
    const int wave = tid >> 5, lane = tid & 31;
    const int wm = wave >> 1, wn = wave & 1;
    const int half = lane >> 4, l16 = lane & 15;

    v8f c[2][4];
#pragma unroll
    for (int i = 0; i < 2; i++)
#pragma unroll
        for (int j = 0; j < 4; j++)
#pragma unroll
            for (int e = 0; e < 8; e++) c[i][j][e] = 0.0f;

    // ---- prologue: stage tile 0 into buffer 0 ----
    if constexpr (sizeof(TIn) == 4) {
#pragma unroll
        for (int q = 0; q < 4; q++) {
            int e = tid + q * 256;
            int r = e >> 3, kq = e & 7;
            float4 f = *(const float4*)&X[(size_t)(m0 + r) * K + kq * 4];
            unsigned lo = (unsigned)to_bf16(f.x) | ((unsigned)to_bf16(f.y) << 16);
            unsigned hi = (unsigned)to_bf16(f.z) | ((unsigned)to_bf16(f.w) << 16);
            *(uint2*)&lds_a[0][r * LDA + kq * 4] = make_uint2(lo, hi);
        }
        if (wave == 0)
            tdm_load_tile_2d(Wt + (size_t)n0 * K, lds_off(lds_b[0]), 32, 128, K, N, K);
    } else {
        if (wave == 0) {
            tdm_load_tile_2d((const unsigned short*)X + (size_t)m0 * K,
                             lds_off(lds_a[0]), 32, 128, K, BATCH * SEQ, K);
            tdm_load_tile_2d(Wt + (size_t)n0 * K, lds_off(lds_b[0]), 32, 128, K, N, K);
        }
    }

    for (int s = 0; s < NSTEP; s++) {
        const int cur = s & 1, nxt = cur ^ 1;
        const int k1 = (s + 1) * 32;

        if (wave == 0) __builtin_amdgcn_s_wait_tensorcnt(0);   // tile s DMA complete
        __syncthreads();                                       // tile s fully staged; buf nxt free

        // prefetch tile s+1: TDM issues now, overlapping the WMMAs below
        if (s + 1 < NSTEP && wave == 0) {
            if constexpr (sizeof(TIn) != 4)
                tdm_load_tile_2d((const unsigned short*)X + (size_t)m0 * K + k1,
                                 lds_off(lds_a[nxt]), 32, 128, K, BATCH * SEQ, K);
            tdm_load_tile_2d(Wt + (size_t)n0 * K + k1, lds_off(lds_b[nxt]), 32, 128, K, N, K);
        }
        // f32 path: fetch next A data to registers (latency hidden behind compute)
        float4 ar[4];
        if constexpr (sizeof(TIn) == 4) {
            if (s + 1 < NSTEP) {
#pragma unroll
                for (int q = 0; q < 4; q++) {
                    int e = tid + q * 256;
                    int r = e >> 3, kq = e & 7;
                    ar[q] = *(const float4*)&X[(size_t)(m0 + r) * K + k1 + kq * 4];
                }
            }
        }

        // ---- compute on buffer cur ----
        FragB16 a[2], b[4];
#pragma unroll
        for (int i = 0; i < 2; i++) {
            int row = wm * 32 + i * 16 + l16;
            const uint4* p = (const uint4*)&lds_a[cur][row * LDA + half * 8];
            a[i].u[0] = p[0]; a[i].u[1] = p[2];   // K {0..7,16..23} (+8 hi half)
        }
#pragma unroll
        for (int j = 0; j < 4; j++) {
            int col = wn * 64 + j * 16 + l16;
            const uint4* p = (const uint4*)&lds_b[cur][col * LDA + half * 8];
            b[j].u[0] = p[0]; b[j].u[1] = p[2];
        }
#pragma unroll
        for (int i = 0; i < 2; i++)
#pragma unroll
            for (int j = 0; j < 4; j++)
                c[i][j] = __builtin_amdgcn_wmma_f32_16x16x32_bf16(
                    false, a[i].v, false, b[j].v, (short)0, c[i][j], false, false);

        // f32 path: convert + store next A tile (visible at next barrier)
        if constexpr (sizeof(TIn) == 4) {
            if (s + 1 < NSTEP) {
#pragma unroll
                for (int q = 0; q < 4; q++) {
                    int e = tid + q * 256;
                    int r = e >> 3, kq = e & 7;
                    unsigned lo = (unsigned)to_bf16(ar[q].x) | ((unsigned)to_bf16(ar[q].y) << 16);
                    unsigned hi = (unsigned)to_bf16(ar[q].z) | ((unsigned)to_bf16(ar[q].w) << 16);
                    *(uint2*)&lds_a[nxt][r * LDA + kq * 4] = make_uint2(lo, hi);
                }
            }
        }
    }

    // epilogue: C layout -> row = e + 8*half, col = l16
#pragma unroll
    for (int i = 0; i < 2; i++)
#pragma unroll
        for (int j = 0; j < 4; j++)
#pragma unroll
            for (int e = 0; e < 8; e++) {
                int m = m0 + wm * 32 + i * 16 + e + 8 * half;
                int n = n0 + wn * 64 + j * 16 + l16;
                float val = c[i][j][e] + bias[n];
                if (OUT_QKV) {
                    int bb = m >> 10, s2 = m & (SEQ - 1);
                    int h = n >> 6,  d = n & (HEAD_DIM - 1);
                    ((unsigned short*)out)[(((size_t)bb * NUM_HEADS + h) * SEQ + s2) * HEAD_DIM + d] =
                        to_bf16(val);
                } else {
                    ((float*)out)[(size_t)m * N + n] = val;
                }
            }
}

// ---------------------------------------------------------------------------
// Flash attention, causal. grid=(S/128, B*H), 8 waves, 16 q-rows per wave.
// Double-buffered V^T tile: next V block loaded to registers before compute,
// scattered to the other buffer after; one barrier per key block.
// ---------------------------------------------------------------------------
__global__ __launch_bounds__(256) void flash_attn(
    const unsigned short* __restrict__ Q, const unsigned short* __restrict__ Kb,
    const unsigned short* __restrict__ Vb, unsigned short* __restrict__ ctx)
{
    constexpr int LDV = 40;
    constexpr float NEG = -3.0e38f;
    __shared__ unsigned short vt[2][64 * LDV];      // V^T tile: [d][key]
    __shared__ unsigned short pst[8 * 16 * LDV];    // per-wave P staging

    const int bh = blockIdx.y, qt = blockIdx.x;
    const int tid = threadIdx.x;
    const int wave = tid >> 5, lane = tid & 31;
    const int half = lane >> 4, l16 = lane & 15;
    const int q_base = qt * 128 + wave * 16;

    const unsigned short* Qh = Q  + (size_t)bh * SEQ * HEAD_DIM;
    const unsigned short* Kh = Kb + (size_t)bh * SEQ * HEAD_DIM;
    const unsigned short* Vh = Vb + (size_t)bh * SEQ * HEAD_DIM;

    FragB16 aQ[2];
#pragma unroll
    for (int f = 0; f < 2; f++) {
        const uint4* p = (const uint4*)&Qh[(size_t)(q_base + l16) * HEAD_DIM + f * 32 + half * 8];
        aQ[f].u[0] = p[0]; aQ[f].u[1] = p[2];
    }

    float m_run[8], l_run[8];
    v8f acc[4];
#pragma unroll
    for (int i = 0; i < 8; i++) { m_run[i] = NEG; l_run[i] = 0.f; }
#pragma unroll
    for (int f = 0; f < 4; f++)
#pragma unroll
        for (int e = 0; e < 8; e++) acc[f][e] = 0.f;

    const int nkb = qt * 4 + 4;
    const int kk = tid >> 3, d0 = (tid & 7) * 8;    // this thread's V slice

    // prologue: stage V tile 0 into buffer 0
    {
        uint4 vv = *(const uint4*)&Vh[(size_t)kk * HEAD_DIM + d0];
        const unsigned short* sv = (const unsigned short*)&vv;
#pragma unroll
        for (int j = 0; j < 8; j++) vt[0][(d0 + j) * LDV + kk] = sv[j];
    }

    for (int kb = 0; kb < nkb; kb++) {
        const int cur = kb & 1, nxt = cur ^ 1;
        __syncthreads();                              // vt[cur] staged; vt[nxt] free

        // prefetch next V block to registers (latency hidden behind compute)
        uint4 vnx;
        if (kb + 1 < nkb) {
            vnx = *(const uint4*)&Vh[(size_t)((kb + 1) * 32 + kk) * HEAD_DIM + d0];
            __builtin_prefetch(&Kh[(size_t)((kb + 1) * 32 + kk) * HEAD_DIM + d0], 0, 0);
        }

        if (kb * 32 <= q_base + 15) {                 // wave-uniform: EXEC full for WMMA
            // scores: 16x32 tile = two 16x16 WMMAs, K-dim = dh (2 chunks of 32)
            v8f sc[2];
#pragma unroll
            for (int j = 0; j < 2; j++) {
#pragma unroll
                for (int e = 0; e < 8; e++) sc[j][e] = 0.f;
#pragma unroll
                for (int cc = 0; cc < 2; cc++) {
                    FragB16 bK;
                    int key = kb * 32 + j * 16 + l16;
                    const uint4* p = (const uint4*)&Kh[(size_t)key * HEAD_DIM + cc * 32 + half * 8];
                    bK.u[0] = p[0]; bK.u[1] = p[2];
                    sc[j] = __builtin_amdgcn_wmma_f32_16x16x32_bf16(
                        false, aQ[cc].v, false, bK.v, (short)0, sc[j], false, false);
                }
            }
            // scale, causal mask, online softmax (16-lane shuffle reductions)
            float alpha[8];
#pragma unroll
            for (int i = 0; i < 8; i++) {
                int qg = q_base + i + 8 * half;
                float tm = NEG;
#pragma unroll
                for (int j = 0; j < 2; j++) {
                    int kg = kb * 32 + j * 16 + l16;
                    float x = sc[j][i] * 0.125f;      // 1/sqrt(64)
                    if (kg > qg) x = NEG;
                    sc[j][i] = x;
                    tm = fmaxf(tm, x);
                }
#pragma unroll
                for (int off = 1; off < 16; off <<= 1) tm = fmaxf(tm, __shfl_xor(tm, off, 32));
                float nm = fmaxf(m_run[i], tm);
                float mc = fmaxf(nm, -1.0e30f);
                alpha[i] = __expf(fmaxf(m_run[i], -1.0e30f) - mc);
                float s = 0.f;
#pragma unroll
                for (int j = 0; j < 2; j++) {
                    float pv = __expf(sc[j][i] - mc);
                    sc[j][i] = pv;
                    s += pv;
                }
#pragma unroll
                for (int off = 1; off < 16; off <<= 1) s += __shfl_xor(s, off, 32);
                m_run[i] = nm;
                l_run[i] = l_run[i] * alpha[i] + s;
            }
            // stage P (C-layout) into LDS, reload in A-layout
            unsigned short* pw = &pst[wave * 16 * LDV];
#pragma unroll
            for (int i = 0; i < 8; i++) {
                int r = i + 8 * half;
                pw[r * LDV + l16]      = to_bf16(sc[0][i]);
                pw[r * LDV + 16 + l16] = to_bf16(sc[1][i]);
            }
            asm volatile("s_wait_dscnt 0" ::: "memory");
            FragB16 aP;
            {
                const uint4* p = (const uint4*)&pw[l16 * LDV + half * 8];
                aP.u[0] = p[0]; aP.u[1] = p[2];
            }
            // rescale accumulators, PV WMMAs
#pragma unroll
            for (int f = 0; f < 4; f++) {
#pragma unroll
                for (int e = 0; e < 8; e++) acc[f][e] *= alpha[e];
                FragB16 bV;
                const uint4* p = (const uint4*)&vt[cur][(f * 16 + l16) * LDV + half * 8];
                bV.u[0] = p[0]; bV.u[1] = p[2];
                acc[f] = __builtin_amdgcn_wmma_f32_16x16x32_bf16(
                    false, aP.v, false, bV.v, (short)0, acc[f], false, false);
            }
        }

        // scatter next V block into the other buffer (visible at next barrier)
        if (kb + 1 < nkb) {
            const unsigned short* sv = (const unsigned short*)&vnx;
#pragma unroll
            for (int j = 0; j < 8; j++) vt[nxt][(d0 + j) * LDV + kk] = sv[j];
        }
    }

    // normalize and store context (bf16, merged-head [B,S,D])
    const int bb = bh >> 4, h = bh & 15;
#pragma unroll
    for (int i = 0; i < 8; i++) {
        float inv = l_run[i] > 0.f ? 1.f / l_run[i] : 0.f;
        int qg = q_base + i + 8 * half;
#pragma unroll
        for (int f = 0; f < 4; f++) {
            int d = f * 16 + l16;
            ctx[((size_t)bb * SEQ + qg) * D_MODEL + h * HEAD_DIM + d] = to_bf16(acc[f][i] * inv);
        }
    }
}

// ---------------------------------------------------------------------------
extern "C" void kernel_launch(void* const* d_in, const int* in_sizes, int n_in,
                              void* d_out, int out_size, void* d_ws, size_t ws_size,
                              hipStream_t stream) {
    const float* xq = (const float*)d_in[0];
    const float* xk = (const float*)d_in[1];
    const float* xv = (const float*)d_in[2];
    const float* Wq = (const float*)d_in[3];
    const float* bq = (const float*)d_in[4];
    const float* Wk = (const float*)d_in[5];
    const float* bk = (const float*)d_in[6];
    const float* Wv = (const float*)d_in[7];
    const float* bv = (const float*)d_in[8];
    const float* Wo = (const float*)d_in[9];
    const float* bo = (const float*)d_in[10];

    const size_t elems = (size_t)BATCH * SEQ * D_MODEL;      // 8M
    const size_t welems = (size_t)D_MODEL * D_MODEL;         // 1M
    unsigned short* Qb  = (unsigned short*)d_ws;
    unsigned short* Kb  = Qb  + elems;
    unsigned short* Vb  = Kb  + elems;
    unsigned short* Cb  = Vb  + elems;
    unsigned short* Wtq = Cb  + elems;
    unsigned short* Wtk = Wtq + welems;
    unsigned short* Wtv = Wtk + welems;
    unsigned short* Wto = Wtv + welems;

    dim3 blk(256);
    dim3 gT(D_MODEL / 32, D_MODEL / 32);                     // 32 x 32
    transpose_w_bf16<<<gT, blk, 0, stream>>>(Wq, Wtq);
    transpose_w_bf16<<<gT, blk, 0, stream>>>(Wk, Wtk);
    transpose_w_bf16<<<gT, blk, 0, stream>>>(Wv, Wtv);
    transpose_w_bf16<<<gT, blk, 0, stream>>>(Wo, Wto);

    dim3 gGemm(BATCH * SEQ / 128, D_MODEL / 128);            // 64 x 8
    gemm_bf16_wmma<float, 1><<<gGemm, blk, 0, stream>>>(xq, Wtq, bq, Qb);
    gemm_bf16_wmma<float, 1><<<gGemm, blk, 0, stream>>>(xk, Wtk, bk, Kb);
    gemm_bf16_wmma<float, 1><<<gGemm, blk, 0, stream>>>(xv, Wtv, bv, Vb);

    dim3 gAttn(SEQ / 128, BATCH * NUM_HEADS);                // 8 x 128
    flash_attn<<<gAttn, blk, 0, stream>>>(Qb, Kb, Vb, Cb);

    gemm_bf16_wmma<unsigned short, 0><<<gGemm, blk, 0, stream>>>(Cb, Wto, bo, (float*)d_out);
}
